// Classifier_18382460027092
// MI455X (gfx1250) — matrix operand order repaired and checked by reading, not compiled, not run on hardware
//
#include <hip/hip_runtime.h>
#include <hip/hip_bf16.h>
#include <math.h>

// Problem constants (match reference)
#define B_      64
#define C_      1280
#define HW_     49
#define A_      40
#define KBLK    32
#define NBLK    128
#define NCHUNK  (C_ / NBLK)     // 10
#define LDSTRIDE 40             // ushorts per n-row in LDS: 32 data + 8 pad (80 B, 16B-mult)

typedef __attribute__((ext_vector_type(16))) __bf16      v16bf;
typedef __attribute__((ext_vector_type(8)))  float       v8f;
typedef __attribute__((ext_vector_type(4)))  unsigned int uint4v;

union FragU { uint4v u[2]; v16bf v; };

__device__ __forceinline__ unsigned short f2bf(float f) {
    unsigned u = __float_as_uint(f);
    unsigned r = (u + 0x7FFFu + ((u >> 16) & 1u)) >> 16;  // round-to-nearest-even
    return (unsigned short)r;
}

#if __has_builtin(__builtin_amdgcn_cvt_pk_bf16_f32)
typedef __attribute__((ext_vector_type(2))) __bf16 v2bf;
__device__ __forceinline__ unsigned pack_bf(float lo, float hi) {
    union { v2bf v; unsigned u; } cvt;
    cvt.v = __builtin_amdgcn_cvt_pk_bf16_f32(lo, hi);   // one v_cvt_pk_bf16_f32
    return cvt.u;
}
#else
// 3-VALU pack: bias to round-to-nearest, then one v_perm_b32 grabs both
// high halves. perm selector: result bytes 0-1 = S1 bytes 2-3 (sel 0x02,0x03),
// result bytes 2-3 = S0 bytes 2-3 (sel 0x06,0x07).
__device__ __forceinline__ unsigned pack_bf(float lo, float hi) {
    unsigned ul = __float_as_uint(lo) + 0x8000u;
    unsigned uh = __float_as_uint(hi) + 0x8000u;
    return __builtin_amdgcn_perm(uh, ul, 0x07060302u);  // {uh[31:16], ul[31:16]}
}
#endif

__device__ __forceinline__ float hswish(float v) {
    float t = fminf(fmaxf(v + 3.0f, 0.0f), 6.0f);
    return v * t * 0.16666667f;
}

// ---------------------------------------------------------------------------
// Kernel 1: adaptive avg-pool (7x7 -> 1) + h_swish, emit bf16 P[b][c] row-major
// One wave per (b,c) row; 49 contiguous floats per row.
// ---------------------------------------------------------------------------
__global__ void pool_hswish_kernel(const float* __restrict__ x,
                                   unsigned short* __restrict__ p) {
    int gid  = blockIdx.x * blockDim.x + threadIdx.x;
    int row  = gid >> 5;          // (b*C + c)
    int lane = gid & 31;
    if (row >= B_ * C_) return;
    const float* xr = x + (size_t)row * HW_;
    float s = xr[lane];
    if (lane < HW_ - 32) s += xr[lane + 32];
#pragma unroll
    for (int off = 16; off >= 1; off >>= 1) s += __shfl_xor(s, off, 32);
    if (lane == 0) {
        float m = s * (1.0f / 49.0f);
        p[row] = f2bf(hswish(m));   // one-time, keep exact RNE
    }
}

// ---------------------------------------------------------------------------
// W1 staging: each thread owns 4 units of (2 rows x 2 cols) of the 32x128 block
// ---------------------------------------------------------------------------
struct Stage { float2 r0[4]; float2 r1[4]; };

__device__ __forceinline__ void load_stage(const float* __restrict__ W1a,
                                           int kb, int n_off, int tid, Stage& st) {
#pragma unroll
    for (int i = 0; i < 4; ++i) {
        int u  = tid + i * 256;          // 0..1023
        int kp = u >> 6;                 // row pair 0..15
        int np = u & 63;                 // col pair 0..63
        const float* g0 = W1a + (size_t)(kb + kp * 2) * C_ + n_off + np * 2;
        st.r0[i] = *(const float2*)(g0);        // row k,   cols n,n+1
        st.r1[i] = *(const float2*)(g0 + C_);   // row k+1, cols n,n+1
    }
}

__device__ __forceinline__ void store_stage(unsigned short* __restrict__ lds,
                                            int tid, const Stage& st) {
#pragma unroll
    for (int i = 0; i < 4; ++i) {
        int u  = tid + i * 256;
        int kp = u >> 6;
        int np = u & 63;
        int n  = np * 2;
        // transposed [n][k] layout, pairs (k,k+1) packed into one dword
        *(unsigned*)(&lds[n * LDSTRIDE + kp * 2])       = pack_bf(st.r0[i].x, st.r1[i].x);
        *(unsigned*)(&lds[(n + 1) * LDSTRIDE + kp * 2]) = pack_bf(st.r0[i].y, st.r1[i].y);
    }
}

// ---------------------------------------------------------------------------
// Kernel 2: fused per-attribute head GEMM.
//   For (a, n-chunk): H = h_swish(P[64xC] * W1[a][CxC] tile + b1), then
//   partial[b] = sum_d H[b,d] * W2[a,d]  (deterministic, no atomics)
// bf16 WMMA 16x16x32, f32 accumulate, double-buffered LDS software pipeline.
// ---------------------------------------------------------------------------
__global__ void __launch_bounds__(256, 2)
heads_gemm_kernel(const unsigned short* __restrict__ p,     // bf16 P[64][1280]
                  const float* __restrict__ W1,
                  const float* __restrict__ b1,
                  const float* __restrict__ W2,
                  float* __restrict__ partial) {
    __shared__ __align__(16) unsigned short ldsB[2][NBLK * LDSTRIDE];

    const int a     = blockIdx.y;
    const int chunk = blockIdx.x;
    const int n_off = chunk * NBLK;
    const int tid   = threadIdx.x;
    const int lane  = tid & 31;
    const int w     = tid >> 5;     // 8 waves
    const int mt    = w & 3;        // M tile (16 rows each, 4 tiles = 64)
    const int nh    = w >> 2;       // N half (64 cols each)
    const int l16   = lane & 15;

    const float* W1a = W1 + (size_t)a * C_ * C_;

    v8f acc[4] = {};                // 4 N-tiles of 16x16 f32

    // A fragment addressing (ISA 16-bit A 16x32 layout):
    //   lanes<16 : V0-3 = K0..7,  V4-7 = K16..23
    //   lanes>=16: V0-3 = K8..15, V4-7 = K24..31
    const unsigned short* prow = p + (size_t)(mt * 16 + l16) * C_;
    const int k8 = (lane < 16) ? 0 : 8;
    const int kh = (lane < 16) ? 0 : 16;   // unused; B fragment offset is k8

    // Pipeline prologue: stage block 0 into buffer 0
    Stage st;
    load_stage(W1a, 0, n_off, tid, st);
    store_stage(ldsB[0], tid, st);
    __syncthreads();

    for (int kb = 0, it = 0; kb < C_; kb += KBLK, ++it) {
        const unsigned short* cur = ldsB[it & 1];
        unsigned short*       nxt = ldsB[(it & 1) ^ 1];
        const bool more = (kb + KBLK) < C_;

        // Issue next block's global loads first: latency hides behind WMMAs.
        if (more) {
            load_stage(W1a, kb + KBLK, n_off, tid, st);
            if (kb + 2 * KBLK < C_)   // global_prefetch_b8 for block after next
                __builtin_prefetch(W1a + (size_t)(kb + 2 * KBLK + (tid >> 6)) * C_ +
                                       n_off + (tid & 63) * 2, 0, 1);
        }

        // A fragment: two 16B global loads from bf16 P (L2-resident)
        FragU af;
        af.u[0] = *(const uint4v*)(prow + kb + k8);
        af.u[1] = *(const uint4v*)(prow + kb + 16 + k8);

#pragma unroll
        for (int t = 0; t < 4; ++t) {
            // B fragment: col n per lane, K k8..k8+7 and 16+k8..16+k8+7 -> 2x ds_load_b128
            const unsigned short* pb =
                &cur[(nh * 64 + t * 16 + l16) * LDSTRIDE + k8];
            FragU bf;
            bf.u[0] = *(const uint4v*)(pb);
            bf.u[1] = *(const uint4v*)(pb + 16);
            acc[t] = __builtin_amdgcn_wmma_f32_16x16x32_bf16(
                /*neg_a=*/false, af.v, /*neg_b=*/false, bf.v,
                /*c_mod=*/(short)0, acc[t], /*reuse_a=*/false, /*reuse_b=*/false);
        }

        // Convert + store the staged block into the other buffer.
        if (more) store_stage(nxt, tid, st);
        __syncthreads();
    }

    // Epilogue: + b1, h_swish, dot with W2 over this wave's 64 d-columns.
    // C/D layout: VGPR r, lane L -> row m = r + (L<16?0:8), col = L%16.
    float s[8];
#pragma unroll
    for (int r = 0; r < 8; ++r) s[r] = 0.0f;
#pragma unroll
    for (int t = 0; t < 4; ++t) {
        int d = n_off + nh * 64 + t * 16 + l16;
        float b1v = b1[(size_t)a * C_ + d];
        float w2v = W2[(size_t)a * C_ + d];
#pragma unroll
        for (int r = 0; r < 8; ++r) {
            float h = hswish(acc[t][r] + b1v);
            s[r] += h * w2v;
        }
    }
    // Reduce each row-sum across the 16-lane column group (xor 1,2,4,8 stays
    // within each half); lanes 0 and 16 hold totals for rows r and r+8.
#pragma unroll
    for (int r = 0; r < 8; ++r) {
#pragma unroll
        for (int off = 8; off >= 1; off >>= 1) s[r] += __shfl_xor(s[r], off, 32);
        if (l16 == 0) {
            int m = mt * 16 + r + ((lane >= 16) ? 8 : 0);
            partial[(((size_t)a * NCHUNK + chunk) * 2 + nh) * 64 + m] = s[r];
        }
    }
}

// ---------------------------------------------------------------------------
// Kernel 3: out[b,a] = sigmoid(b2[a] + sum over 20 partials)
// ---------------------------------------------------------------------------
__global__ void finalize_kernel(const float* __restrict__ partial,
                                const float* __restrict__ b2,
                                float* __restrict__ out) {
    int idx = blockIdx.x * blockDim.x + threadIdx.x;
    if (idx >= B_ * A_) return;
    int b = idx / A_;
    int a = idx % A_;
    float s = b2[a];
#pragma unroll
    for (int c = 0; c < NCHUNK * 2; ++c)
        s += partial[((size_t)a * NCHUNK * 2 + c) * 64 + b];
    out[idx] = 1.0f / (1.0f + __expf(-s));
}

// ---------------------------------------------------------------------------
extern "C" void kernel_launch(void* const* d_in, const int* in_sizes, int n_in,
                              void* d_out, int out_size, void* d_ws, size_t ws_size,
                              hipStream_t stream) {
    const float* x  = (const float*)d_in[0];   // [64,1280,7,7]
    const float* W1 = (const float*)d_in[1];   // [40,1280,1280]
    const float* b1 = (const float*)d_in[2];   // [40,1280]
    const float* W2 = (const float*)d_in[3];   // [40,1280]
    const float* b2 = (const float*)d_in[4];   // [40]
    float* out = (float*)d_out;                // [64,40]

    unsigned short* p = (unsigned short*)d_ws;                    // 163840 B bf16 P
    float* partial = (float*)((char*)d_ws + 163840);              // 204800 B partials

    pool_hswish_kernel<<<(B_ * C_ * 32 + 255) / 256, 256, 0, stream>>>(x, p);

    dim3 grid2(NCHUNK, A_);
    heads_gemm_kernel<<<grid2, 256, 0, stream>>>(p, W1, b1, W2, partial);

    finalize_kernel<<<(B_ * A_ + 255) / 256, 256, 0, stream>>>(partial, b2, out);
}
